// MultiHeadAttentionWithKVCache_71734543778299
// MI455X (gfx1250) — compile-verified, matrix operand order checked by reading, not run on hardware
//
#include <hip/hip_runtime.h>

typedef __attribute__((ext_vector_type(8)))  __bf16 v8bf;
typedef __attribute__((ext_vector_type(16))) __bf16 v16bf;
typedef __attribute__((ext_vector_type(8)))  float  v8f;

#define ZERO8F (v8f{0.f,0.f,0.f,0.f,0.f,0.f,0.f,0.f})

__device__ __forceinline__ v8bf ld8(const __bf16* p) {
  return *reinterpret_cast<const v8bf*>(p);
}
__device__ __forceinline__ v16bf cat16(v8bf lo, v8bf hi) {
  return __builtin_shufflevector(lo, hi, 0,1,2,3,4,5,6,7,8,9,10,11,12,13,14,15);
}
__device__ __forceinline__ v8f wmma_bf16(v16bf a, v16bf b, v8f c) {
  // (neg_a, A, neg_b, B, c_mod, C, reuse_a, reuse_b)
  return __builtin_amdgcn_wmma_f32_16x16x32_bf16(false, a, false, b, (short)0, c,
                                                 false, false);
}
__device__ __forceinline__ float redmax16(float v) {
  v = fmaxf(v, __shfl_xor(v, 1, 32));
  v = fmaxf(v, __shfl_xor(v, 2, 32));
  v = fmaxf(v, __shfl_xor(v, 4, 32));
  v = fmaxf(v, __shfl_xor(v, 8, 32));
  return v;
}
__device__ __forceinline__ float redsum16(float v) {
  v += __shfl_xor(v, 1, 32);
  v += __shfl_xor(v, 2, 32);
  v += __shfl_xor(v, 4, 32);
  v += __shfl_xor(v, 8, 32);
  return v;
}

// ---------------------------------------------------------------------------
// elementwise f32 -> bf16
__global__ __launch_bounds__(256) void cvt_kernel(const float* __restrict__ in,
                                                  __bf16* __restrict__ out, int n) {
  int i = blockIdx.x * 256 + threadIdx.x;
  if (i < n) out[i] = (__bf16)in[i];
}

// copy past K/V into f32 cache outputs + bf16 (K row-major, V transposed [B,H,HD,T])
__global__ __launch_bounds__(256) void copy_past_kernel(
    const float* __restrict__ pk, const float* __restrict__ pv,
    float* __restrict__ ok, float* __restrict__ ov,
    __bf16* __restrict__ kbf, __bf16* __restrict__ vtb) {
  int i = blockIdx.x * 256 + threadIdx.x;
  if (i >= 2 * 16 * 1024 * 128) return;
  const int hd = i & 127, p = (i >> 7) & 1023, h = (i >> 17) & 15, b = i >> 21;
  const int bh = b * 16 + h;
  const size_t fidx = ((size_t)(bh * 2048 + p)) * 128 + hd;
  const float kv = pk[i], vv = pv[i];
  ok[fidx] = kv;
  kbf[fidx] = (__bf16)kv;
  ov[fidx] = vv;
  vtb[((size_t)(bh * 128 + hd)) * 2048 + p] = (__bf16)vv;
}

// ---------------------------------------------------------------------------
// WMMA GEMM: C[M=2048,N=2048] = A_bf16[M,K=2048] x W_bf16[N,K]^T + bias
// wave computes 16x128 tile (8 WMMA accumulators).
// mode 0: Q  -> bf16 [B,H,S,HD]
// mode 1: K  -> f32 cache [B,H,T,HD] (rows P..) + bf16 same layout
// mode 2: V  -> f32 cache [B,H,T,HD] (rows P..) + bf16 transposed [B,H,HD,T]
// mode 3: O  -> f32 [B,S,D]
__global__ __launch_bounds__(256) void gemm16_kernel(
    const __bf16* __restrict__ A, const __bf16* __restrict__ W,
    const float* __restrict__ bias, float* __restrict__ outF,
    __bf16* __restrict__ outB, int mode) {
  constexpr int K = 2048;
  const int wid  = blockIdx.x * (blockDim.x >> 5) + (threadIdx.x >> 5);
  const int lane = threadIdx.x & 31;
  const int l15 = lane & 15, g = lane >> 4;
  const int mt = wid >> 4;          // 0..127  (2048/16 m-tiles)
  const int ng = wid & 15;          // 0..15   (2048/128 n-groups)
  const int m0 = mt * 16, n0 = ng * 128;
  const __bf16* arow = A + (size_t)(m0 + l15) * K;

  v8f acc[8];
#pragma unroll
  for (int t = 0; t < 8; ++t) acc[t] = ZERO8F;

  for (int ks = 0; ks < K / 32; ++ks) {
    const int k0 = ks * 32;
    v16bf af = cat16(ld8(arow + k0 + g * 8), ld8(arow + k0 + 16 + g * 8));
#pragma unroll
    for (int nt = 0; nt < 8; ++nt) {
      const __bf16* wrow = W + (size_t)(n0 + nt * 16 + l15) * K;
      v16bf bfr = cat16(ld8(wrow + k0 + g * 8), ld8(wrow + k0 + 16 + g * 8));
      acc[nt] = wmma_bf16(af, bfr, acc[nt]);
    }
  }

#pragma unroll
  for (int nt = 0; nt < 8; ++nt) {
    const int c = n0 + nt * 16 + l15;
    const float bv = bias[c];
#pragma unroll
    for (int r = 0; r < 8; ++r) {
      const int m = m0 + r + 8 * g;   // C layout: VGPR r -> row r + 8*(lane/16)
      const float val = acc[nt][r] + bv;
      if (mode == 3) {
        outF[(size_t)m * 2048 + c] = val;
      } else {
        const int b = m >> 10, s = m & 1023;
        const int h = c >> 7, hd = c & 127;
        const int bh = b * 16 + h;
        if (mode == 0) {
          outB[((size_t)(bh * 1024 + s)) * 128 + hd] = (__bf16)val;
        } else {
          const int t = 1024 + s;
          const size_t fidx = ((size_t)(bh * 2048 + t)) * 128 + hd;
          outF[fidx] = val;
          if (mode == 1)
            outB[fidx] = (__bf16)val;
          else
            outB[((size_t)(bh * 128 + hd)) * 2048 + t] = (__bf16)val;
        }
      }
    }
  }
}

// ---------------------------------------------------------------------------
// Flash attention: 1 wave per (b,h, 16-row q tile). Online softmax over
// 32-key chunks, QK^T and PV via WMMA, P transposed C->A layout via LDS.
__global__ __launch_bounds__(256) void attn_kernel(
    const __bf16* __restrict__ qb, const __bf16* __restrict__ kb,
    const __bf16* __restrict__ vt, __bf16* __restrict__ ob) {
  __shared__ __bf16 plds[8][16 * 32];
  const int wave = threadIdx.x >> 5, lane = threadIdx.x & 31;
  const int l15 = lane & 15, g = lane >> 4;
  const int job = blockIdx.x * 8 + wave;  // 0..2047 = B*H*(S/16)
  const int qt = job & 63, bh = job >> 6;
  const int i0 = qt * 16;
  constexpr float SCALE = 0.08838834764831845f;  // 1/sqrt(128)

  const __bf16* qrow = qb + ((size_t)(bh * 1024 + i0 + l15)) * 128;
  v16bf aq[4];
#pragma unroll
  for (int ks = 0; ks < 4; ++ks)
    aq[ks] = cat16(ld8(qrow + ks * 32 + g * 8), ld8(qrow + ks * 32 + 16 + g * 8));

  v8f acc[8];
#pragma unroll
  for (int t = 0; t < 8; ++t) acc[t] = ZERO8F;
  float mrow[8], lrow[8];
#pragma unroll
  for (int r = 0; r < 8; ++r) { mrow[r] = -3.0e38f; lrow[r] = 0.f; }

  const __bf16* kbase = kb + (size_t)bh * 2048 * 128;
  const __bf16* vbase = vt + (size_t)bh * 128 * 2048;
  const int lastKey = min(1024 + i0 + 15, 1919);  // causal & pad (T-128)

  for (int cb = 0; cb <= lastKey; cb += 32) {
    v8f st0 = ZERO8F, st1 = ZERO8F;
    const __bf16* kr0 = kbase + (size_t)(cb + l15) * 128;
    const __bf16* kr1 = kr0 + 16 * 128;
#pragma unroll
    for (int ks = 0; ks < 4; ++ks) {
      st0 = wmma_bf16(aq[ks],
                      cat16(ld8(kr0 + ks * 32 + g * 8), ld8(kr0 + ks * 32 + 16 + g * 8)),
                      st0);
      st1 = wmma_bf16(aq[ks],
                      cat16(ld8(kr1 + ks * 32 + g * 8), ld8(kr1 + ks * 32 + 16 + g * 8)),
                      st1);
    }
    const int j0 = cb + l15, j1 = j0 + 16;
    float alpha[8];
#pragma unroll
    for (int r = 0; r < 8; ++r) {
      const int pos = 1024 + i0 + r + 8 * g;  // global query position
      float s0 = st0[r] * SCALE + ((j0 > pos || j0 >= 1920) ? -1e9f : 0.f);
      float s1 = st1[r] * SCALE + ((j1 > pos || j1 >= 1920) ? -1e9f : 0.f);
      float mnew = fmaxf(mrow[r], redmax16(fmaxf(s0, s1)));
      float a = __expf(mrow[r] - mnew);
      float p0 = __expf(s0 - mnew), p1 = __expf(s1 - mnew);
      lrow[r] = lrow[r] * a + redsum16(p0 + p1);
      mrow[r] = mnew;
      alpha[r] = a;
      plds[wave][(r + 8 * g) * 32 + l15]      = (__bf16)p0;
      plds[wave][(r + 8 * g) * 32 + 16 + l15] = (__bf16)p1;
    }
#pragma unroll
    for (int nt = 0; nt < 8; ++nt)
#pragma unroll
      for (int r = 0; r < 8; ++r) acc[nt][r] *= alpha[r];

    asm volatile("s_wait_dscnt 0x0" ::: "memory");  // per-wave DS in-order + wait
    v16bf pf = cat16(ld8(&plds[wave][l15 * 32 + g * 8]),
                     ld8(&plds[wave][l15 * 32 + 16 + g * 8]));
#pragma unroll
    for (int nt = 0; nt < 8; ++nt) {
      const __bf16* vr = vbase + (size_t)(nt * 16 + l15) * 2048 + cb;
      acc[nt] = wmma_bf16(pf, cat16(ld8(vr + g * 8), ld8(vr + 16 + g * 8)), acc[nt]);
    }
  }

  const int b = bh >> 4, h = bh & 15;
#pragma unroll
  for (int r = 0; r < 8; ++r) {
    const float inv = 1.0f / lrow[r];
    const int i = i0 + r + 8 * g;
#pragma unroll
    for (int nt = 0; nt < 8; ++nt)
      ob[((size_t)(b * 1024 + i)) * 2048 + h * 128 + nt * 16 + l15] =
          (__bf16)(acc[nt][r] * inv);
  }
}

// ---------------------------------------------------------------------------
extern "C" void kernel_launch(void* const* d_in, const int* in_sizes, int n_in,
                              void* d_out, int out_size, void* d_ws, size_t ws_size,
                              hipStream_t stream) {
  (void)in_sizes; (void)n_in; (void)out_size; (void)ws_size;
  const float* hidden = (const float*)d_in[0];
  const float* past_k = (const float*)d_in[1];
  const float* past_v = (const float*)d_in[2];
  const float* Wq = (const float*)d_in[5];
  const float* bq = (const float*)d_in[6];
  const float* Wk = (const float*)d_in[7];
  const float* bk = (const float*)d_in[8];
  const float* Wv = (const float*)d_in[9];
  const float* bv = (const float*)d_in[10];
  const float* Wo = (const float*)d_in[11];
  const float* bo = (const float*)d_in[12];

  float* out   = (float*)d_out;
  float* out_k = out + (size_t)2 * 1024 * 2048;        // after output [B,S,D]
  float* out_v = out_k + (size_t)2 * 16 * 2048 * 128;  // after k [B,H,T,HD]

  char* ws = (char*)d_ws;
  const size_t nHid = (size_t)2 * 1024 * 2048;    // 4194304
  const size_t nW   = (size_t)2048 * 2048;        // 4194304
  const size_t nKV  = (size_t)2 * 16 * 2048 * 128;// 8388608
  __bf16* hB  = (__bf16*)ws; ws += nHid * 2;
  __bf16* WqB = (__bf16*)ws; ws += nW * 2;
  __bf16* WkB = (__bf16*)ws; ws += nW * 2;
  __bf16* WvB = (__bf16*)ws; ws += nW * 2;
  __bf16* WoB = (__bf16*)ws; ws += nW * 2;
  __bf16* qB  = (__bf16*)ws; ws += nHid * 2;
  __bf16* kB  = (__bf16*)ws; ws += nKV * 2;
  __bf16* vTB = (__bf16*)ws; ws += nKV * 2;
  __bf16* aB  = (__bf16*)ws; ws += nHid * 2;

  dim3 blk(256);
  cvt_kernel<<<(unsigned)((nHid + 255) / 256), blk, 0, stream>>>(hidden, hB, (int)nHid);
  cvt_kernel<<<(unsigned)((nW + 255) / 256), blk, 0, stream>>>(Wq, WqB, (int)nW);
  cvt_kernel<<<(unsigned)((nW + 255) / 256), blk, 0, stream>>>(Wk, WkB, (int)nW);
  cvt_kernel<<<(unsigned)((nW + 255) / 256), blk, 0, stream>>>(Wv, WvB, (int)nW);
  cvt_kernel<<<(unsigned)((nW + 255) / 256), blk, 0, stream>>>(Wo, WoB, (int)nW);

  const size_t nPast = (size_t)2 * 16 * 1024 * 128;
  copy_past_kernel<<<(unsigned)((nPast + 255) / 256), blk, 0, stream>>>(
      past_k, past_v, out_k, out_v, kB, vTB);

  // Q, K, V projections (2048 wave-jobs each -> 256 blocks of 8 waves)
  gemm16_kernel<<<256, blk, 0, stream>>>(hB, WqB, bq, nullptr, qB, 0);
  gemm16_kernel<<<256, blk, 0, stream>>>(hB, WkB, bk, out_k, kB, 1);
  gemm16_kernel<<<256, blk, 0, stream>>>(hB, WvB, bv, out_v, vTB, 2);

  // flash attention (2048 wave-jobs -> 256 blocks)
  attn_kernel<<<256, blk, 0, stream>>>(qB, kB, vTB, aB);

  // output projection
  gemm16_kernel<<<256, blk, 0, stream>>>(aB, WoB, bo, out, nullptr, 3);
}
